// Decoder_26474178413107
// MI455X (gfx1250) — matrix operand order repaired and checked by reading, not compiled, not run on hardware
//
#include <hip/hip_runtime.h>

// ---------------------------------------------------------------------------
// Problem constants (match reference)
// ---------------------------------------------------------------------------
constexpr int kV  = 32000;
constexpr int kD  = 1024;
constexpr int kH  = 16;
constexpr int kL  = 4;
constexpr int kB  = 2;
constexpr int kS  = 512;
constexpr int kSE = 512;
constexpr int kDH = 64;      // D / H
constexpr int kF  = 4 * kD;  // FFN hidden
constexpr int kBS = kB * kS; // 1024 rows of activations
constexpr float kEPS = 1e-3f;

// ---------------------------------------------------------------------------
// Types
// ---------------------------------------------------------------------------
typedef __attribute__((ext_vector_type(16))) _Float16 v16h;
typedef __attribute__((ext_vector_type(8)))  _Float16 v8h;
typedef __attribute__((ext_vector_type(8)))  float    v8f;
typedef __attribute__((ext_vector_type(4)))  unsigned int u32x4;
typedef __attribute__((ext_vector_type(8)))  int          i32x8;
typedef __attribute__((ext_vector_type(4)))  int          i32x4;

union HFrag { v16h v; v8h h[2]; };

#define TB   256   // 8 wave32 waves
#define BM   64
#define BN   64
#define BK   32
#define LSTR 40    // padded LDS row stride in halves (rows stay 16B aligned)

#define FLAG_RELU   1
#define FLAG_TRANSB 2   // B given as [N,K] row-major
#define FLAG_BF16   4   // B operand is f16 (else fp32, converted while staging)
#define FLAG_OUTF16 8   // C written as f16 (else fp32)

// ---------------------------------------------------------------------------
// Tensor Data Mover support (CDNA5). Guarded: falls back to manual staging.
// ---------------------------------------------------------------------------
#if defined(__has_builtin)
#if __has_builtin(__builtin_amdgcn_tensor_load_to_lds) && \
    __has_builtin(__builtin_amdgcn_s_wait_tensorcnt)
#define HAVE_TDM 1
#endif
#endif
#ifndef HAVE_TDM
#define HAVE_TDM 0
#endif

#if HAVE_TDM
// 2-D f16 tile load Global->LDS via TDM (D# per ISA ch.8).
// tile_d0 halves per row, tile_d1 rows, stride in halves; pads LDS with
// 4 DWORDs every 16 DWORDs -> 40-half padded rows (matches LSTR).
__device__ __forceinline__ void tdm_load_tile_f16(unsigned lds_off,
                                                  unsigned long long gaddr,
                                                  unsigned tile_d0,
                                                  unsigned tile_d1,
                                                  unsigned long long stride_h) {
  u32x4 g0;
  g0[0] = 1u;                                      // count=1, user desc
  g0[1] = lds_off;                                 // lds_addr
  g0[2] = (unsigned)(gaddr & 0xFFFFFFFFu);         // global_addr[31:0]
  g0[3] = (unsigned)((gaddr >> 32) & 0x1FFFFFFu)   // global_addr[56:32]
          | (2u << 30);                            // type=2 ("image")
  i32x8 g1;
  // data_size=1 (2B), pad_enable, pad_interval=3 (16 DW), pad_amount=3 (4 DW)
  g1[0] = (int)((1u << 16) | (1u << 20) | (3u << 22) | (3u << 25));
  const unsigned td0 = 0x40000000u;                // huge tensor dims: no OOB clip
  const unsigned td1 = 0x40000000u;
  g1[1] = (int)((td0 & 0xFFFFu) << 16);                             // dim0 lo16
  g1[2] = (int)(((td0 >> 16) & 0xFFFFu) | ((td1 & 0xFFFFu) << 16)); // dim0 hi, dim1 lo
  g1[3] = (int)(((td1 >> 16) & 0xFFFFu) | ((tile_d0 & 0xFFFFu) << 16));
  g1[4] = (int)(tile_d1 & 0xFFFFu);                // tile_dim1; tile_dim2=0
  g1[5] = (int)(unsigned)(stride_h & 0xFFFFFFFFu); // dim0_stride lo32
  g1[6] = (int)(unsigned)((stride_h >> 32) & 0xFFFFu);
  g1[7] = 0;
  const i32x4 zero4 = {0, 0, 0, 0};
  const i32x8 zero8 = {0, 0, 0, 0, 0, 0, 0, 0};
  // amdgpu-toolchain (clang-23) 6-arg form:
  // (u32x4 g0, i32x8 g1, i32x4 g2, i32x4 g3, i32x8, i32 cpol)
  __builtin_amdgcn_tensor_load_to_lds(g0, g1, zero4, zero4, zero8, 0);
}
#endif

// ---------------------------------------------------------------------------
// Generic batched GEMM:  C[z] = alpha * A[z] (MxK) * B[z] (KxN) + bias
// A: f16 [M,K] row-major.  B: fp32 [K,N] | f16 [K,N] | f16 [N,K] (TRANSB).
// Math: v_wmma_f32_16x16x32_f16, fp32 accumulation.
// Batch offsets (in elements of the respective buffer):
//   off = (z / bmod) * s?1 + (z % bmod) * s?2
// ---------------------------------------------------------------------------
__global__ __launch_bounds__(TB)
void gemm_wmma(const void* __restrict__ Av, const void* __restrict__ Bv,
               void* __restrict__ Cv, const float* __restrict__ bias,
               int M, int N, int K, int lda, int ldb, int ldc,
               int bmod, long sA1, long sA2, long sB1, long sB2,
               long sC1, long sC2, long sBias, float alpha, int flags) {
  (void)M; (void)N;
  const int z  = blockIdx.z;
  const int zq = z / bmod;
  const int zr = z % bmod;
  const _Float16* A = (const _Float16*)Av + zq * sA1 + zr * sA2;
  const char* B8 = (const char*)Bv +
                   (zq * sB1 + zr * sB2) * ((flags & FLAG_BF16) ? 2 : 4);
  char* C8 = (char*)Cv + (zq * sC1 + zr * sC2) * ((flags & FLAG_OUTF16) ? 2 : 4);
  if (bias) bias += zr * sBias;

  __shared__ __align__(16) _Float16 Ash[BM * LSTR];
  __shared__ __align__(16) _Float16 Bsh[BN * LSTR];

  const int t    = threadIdx.x;
  const int lane = t & 31;
  const int wave = t >> 5;
  const int wm   = wave & 3;          // M-tile 0..3
  const int wn   = (wave >> 2) << 1;  // N-tile base 0 or 2
  const int lm   = lane & 15;
  const int lh   = lane >> 4;         // 0 or 1
  const int m0   = blockIdx.y * BM;
  const int n0   = blockIdx.x * BN;

  v8f acc0 = {0.f, 0.f, 0.f, 0.f, 0.f, 0.f, 0.f, 0.f};
  v8f acc1 = {0.f, 0.f, 0.f, 0.f, 0.f, 0.f, 0.f, 0.f};

  for (int k0 = 0; k0 < K; k0 += BK) {
    // ---------------- stage A tile (BM x BK halves) ----------------
#if HAVE_TDM
    if (wave == 0) {
      tdm_load_tile_f16((unsigned)(unsigned long long)(void*)Ash,
                        (unsigned long long)(const void*)(A + (long)m0 * lda + k0),
                        BK, BM, (unsigned long long)lda);
    }
#else
    {
      const int row = t >> 2;             // 0..63
      const int col = (t & 3) << 3;       // 0,8,16,24 halves
      *(v8h*)&Ash[row * LSTR + col] =
          *(const v8h*)(A + (long)(m0 + row) * lda + k0 + col);
    }
#endif
    // ---------------- stage B tile into Bsh[n][k] ----------------
    if (flags & FLAG_TRANSB) {            // f16 [N,K]: direct tile copy
#if HAVE_TDM
      if (wave == 1) {
        const _Float16* Bh = (const _Float16*)B8;
        tdm_load_tile_f16((unsigned)(unsigned long long)(void*)Bsh,
                          (unsigned long long)(const void*)(Bh + (long)n0 * ldb + k0),
                          BK, BN, (unsigned long long)ldb);
      }
#else
      const int row = t >> 2;
      const int col = (t & 3) << 3;
      const _Float16* Bh = (const _Float16*)B8;
      *(v8h*)&Bsh[row * LSTR + col] =
          *(const v8h*)(Bh + (long)(n0 + row) * ldb + k0 + col);
#endif
    } else if (flags & FLAG_BF16) {       // f16 [K,N]: transpose while staging
      const int kk = t >> 3;              // 0..31
      const int nn = (t & 7) << 3;        // 0..56
      const _Float16* Bh = (const _Float16*)B8;
      const v8h hv = *(const v8h*)(Bh + (long)(k0 + kk) * ldb + n0 + nn);
#pragma unroll
      for (int j = 0; j < 8; ++j) Bsh[(nn + j) * LSTR + kk] = hv[j];
    } else {                              // fp32 [K,N]: convert + transpose
      const int kk = t >> 3;
      const int nn = (t & 7) << 3;
      const float* Bf = (const float*)B8;
      const float* src = Bf + (long)(k0 + kk) * ldb + n0 + nn;
      const float4 f0 = *(const float4*)(src);
      const float4 f1 = *(const float4*)(src + 4);
      Bsh[(nn + 0) * LSTR + kk] = (_Float16)f0.x;
      Bsh[(nn + 1) * LSTR + kk] = (_Float16)f0.y;
      Bsh[(nn + 2) * LSTR + kk] = (_Float16)f0.z;
      Bsh[(nn + 3) * LSTR + kk] = (_Float16)f0.w;
      Bsh[(nn + 4) * LSTR + kk] = (_Float16)f1.x;
      Bsh[(nn + 5) * LSTR + kk] = (_Float16)f1.y;
      Bsh[(nn + 6) * LSTR + kk] = (_Float16)f1.z;
      Bsh[(nn + 7) * LSTR + kk] = (_Float16)f1.w;
    }
#if HAVE_TDM
    __builtin_amdgcn_s_wait_tensorcnt(0);
#endif
    __syncthreads();

    // ---------------- fragments (contiguous k = 16*lh + s, A and B alike) ----------------
    HFrag af, bf0, bf1;
    const _Float16* ap = &Ash[(wm * 16 + lm) * LSTR + (lh << 4)];
    af.h[0] = *(const v8h*)(ap);
    af.h[1] = *(const v8h*)(ap + 8);
    const _Float16* bp0 = &Bsh[((wn + 0) * 16 + lm) * LSTR + (lh << 4)];
    bf0.h[0] = *(const v8h*)(bp0);
    bf0.h[1] = *(const v8h*)(bp0 + 8);
    const _Float16* bp1 = &Bsh[((wn + 1) * 16 + lm) * LSTR + (lh << 4)];
    bf1.h[0] = *(const v8h*)(bp1);
    bf1.h[1] = *(const v8h*)(bp1 + 8);

    acc0 = __builtin_amdgcn_wmma_f32_16x16x32_f16(false, af.v, false, bf0.v,
                                                  (short)0, acc0, false, false);
    acc1 = __builtin_amdgcn_wmma_f32_16x16x32_f16(false, af.v, false, bf1.v,
                                                  (short)0, acc1, false, false);
    __syncthreads();
  }

  // ---------------- epilogue: lane = column lm, rows lh*8 + r ----------------
  const int gm = m0 + wm * 16 + (lh << 3);
#pragma unroll
  for (int tile = 0; tile < 2; ++tile) {
    const int gn = n0 + (wn + tile) * 16 + lm;
    const float bv = bias ? bias[gn] : 0.f;
#pragma unroll
    for (int r = 0; r < 8; ++r) {
      float v = (tile ? acc1[r] : acc0[r]) * alpha + bv;
      if (flags & FLAG_RELU) v = fmaxf(v, 0.f);
      const long idx = (long)(gm + r) * ldc + gn;
      if (flags & FLAG_OUTF16) ((_Float16*)C8)[idx] = (_Float16)v;
      else                     ((float*)C8)[idx] = v;
    }
  }
}

// ---------------------------------------------------------------------------
// Embedding + positional encoding -> fp32 X and f16 copy
// ---------------------------------------------------------------------------
__global__ __launch_bounds__(256)
void embed_kernel(const int* __restrict__ seq, const float* __restrict__ emb,
                  const float* __restrict__ pes, float* __restrict__ x32,
                  void* __restrict__ x16, int total) {
  const int i = blockIdx.x * 256 + threadIdx.x;
  if (i >= total) return;
  const int d   = i % kD;
  const int row = i / kD;
  const int s   = row % kS;
  const float v = emb[(long)seq[row] * kD + d] + pes[s * kD + d];
  x32[i] = v;
  ((_Float16*)x16)[i] = (_Float16)v;
}

// ---------------------------------------------------------------------------
// fp32 -> f16 conversion (encoder_out)
// ---------------------------------------------------------------------------
__global__ __launch_bounds__(256)
void cvt16_kernel(const float* __restrict__ src, void* __restrict__ dst,
                  int total) {
  const int i = blockIdx.x * 256 + threadIdx.x;
  if (i < total) ((_Float16*)dst)[i] = (_Float16)src[i];
}

// ---------------------------------------------------------------------------
// Row softmax, optional strict-causal mask (row q attends to k < q; q==0 row
// becomes all zeros, matching the reference). fp32 scores in, f16 probs out.
// grid: (Sq, B*H), block: 256
// ---------------------------------------------------------------------------
__global__ __launch_bounds__(256)
void softmax_kernel(float* __restrict__ sc, void* __restrict__ prv,
                    int Sq, int Sk, int causal) {
  const int q = blockIdx.x;
  const long base = ((long)blockIdx.y * Sq + q) * Sk;
  float* row = sc + base;
  _Float16* pr = (_Float16*)prv + base;
  const int valid = causal ? q : Sk;
  const int t = threadIdx.x;
  __shared__ float red[8];

  float mx = -3.0e38f;
  for (int i = t; i < valid; i += 256) mx = fmaxf(mx, row[i]);
  for (int off = 16; off > 0; off >>= 1) mx = fmaxf(mx, __shfl_down(mx, off, 32));
  if ((t & 31) == 0) red[t >> 5] = mx;
  __syncthreads();
  if (t == 0) {
    float m = red[0];
    for (int i = 1; i < 8; ++i) m = fmaxf(m, red[i]);
    red[0] = m;
  }
  __syncthreads();
  mx = red[0];
  __syncthreads();

  float sum = 0.f;
  for (int i = t; i < valid; i += 256) {
    const float e = __expf(row[i] - mx);
    row[i] = e;
    sum += e;
  }
  for (int off = 16; off > 0; off >>= 1) sum += __shfl_down(sum, off, 32);
  if ((t & 31) == 0) red[t >> 5] = sum;
  __syncthreads();
  if (t == 0) {
    float s = red[0];
    for (int i = 1; i < 8; ++i) s += red[i];
    red[0] = s;
  }
  __syncthreads();
  sum = red[0];

  const float inv = (valid > 0 && sum > 0.f) ? 1.0f / sum : 0.0f;
  for (int i = t; i < Sk; i += 256) {
    pr[i] = (_Float16)((i < valid) ? row[i] * inv : 0.0f);
  }
}

// ---------------------------------------------------------------------------
// Fused residual + inference BatchNorm; updates fp32 X and its f16 copy.
// ---------------------------------------------------------------------------
__global__ __launch_bounds__(256)
void bnres_kernel(float* __restrict__ x32, void* __restrict__ x16,
                  const float* __restrict__ a, const float* __restrict__ g,
                  const float* __restrict__ be, const float* __restrict__ mu,
                  const float* __restrict__ vr, int total) {
  const int i = blockIdx.x * 256 + threadIdx.x;
  if (i >= total) return;
  const int c = i % kD;
  const float s = a[i] + x32[i];
  const float r = (s - mu[c]) * rsqrtf(vr[c] + kEPS) * g[c] + be[c];
  x32[i] = r;
  ((_Float16*)x16)[i] = (_Float16)r;
}

// ---------------------------------------------------------------------------
// Host-side GEMM launcher
// ---------------------------------------------------------------------------
static inline void launch_gemm(hipStream_t s, const void* A, const void* B,
                               void* C, const float* bias, int M, int N, int K,
                               int lda, int ldb, int ldc, int nz, int bmod,
                               long sA1, long sA2, long sB1, long sB2,
                               long sC1, long sC2, long sBias,
                               float alpha, int flags) {
  dim3 grid(N / BN, M / BM, nz), block(TB);
  gemm_wmma<<<grid, block, 0, s>>>(A, B, C, bias, M, N, K, lda, ldb, ldc,
                                   bmod, sA1, sA2, sB1, sB2, sC1, sC2, sBias,
                                   alpha, flags);
}

extern "C" void kernel_launch(void* const* d_in, const int* in_sizes, int n_in,
                              void* d_out, int out_size, void* d_ws, size_t ws_size,
                              hipStream_t stream) {
  (void)in_sizes; (void)n_in; (void)out_size; (void)ws_size;

  const int*   seq   = (const int*)d_in[0];
  const float* enc   = (const float*)d_in[1];
  const float* pes   = (const float*)d_in[2];
  const float* emb   = (const float*)d_in[3];
  const float* wq_s  = (const float*)d_in[4];
  const float* bq_s  = (const float*)d_in[5];
  const float* wk_s  = (const float*)d_in[6];
  const float* bk_s  = (const float*)d_in[7];
  const float* wv_s  = (const float*)d_in[8];
  const float* bv_s  = (const float*)d_in[9];
  const float* wo_s  = (const float*)d_in[10];
  const float* bo_s  = (const float*)d_in[11];
  const float* wq_c  = (const float*)d_in[12];
  const float* bq_c  = (const float*)d_in[13];
  const float* wk_c  = (const float*)d_in[14];
  const float* bk_c  = (const float*)d_in[15];
  const float* wv_c  = (const float*)d_in[16];
  const float* bv_c  = (const float*)d_in[17];
  const float* wo_c  = (const float*)d_in[18];
  const float* bo_c  = (const float*)d_in[19];
  const float* g1    = (const float*)d_in[20];
  const float* be1   = (const float*)d_in[21];
  const float* m1    = (const float*)d_in[22];
  const float* v1    = (const float*)d_in[23];
  const float* g2    = (const float*)d_in[24];
  const float* be2   = (const float*)d_in[25];
  const float* m2    = (const float*)d_in[26];
  const float* v2    = (const float*)d_in[27];
  const float* g3    = (const float*)d_in[28];
  const float* be3   = (const float*)d_in[29];
  const float* m3    = (const float*)d_in[30];
  const float* v3    = (const float*)d_in[31];
  const float* w1    = (const float*)d_in[32];
  const float* b1    = (const float*)d_in[33];
  const float* w2    = (const float*)d_in[34];
  const float* b2    = (const float*)d_in[35];
  const float* w_out = (const float*)d_in[36];
  const float* b_out = (const float*)d_in[37];

  // Workspace layout (byte offsets). Activation rows: [B*S, D] = [1024, 1024].
  const size_t NM = (size_t)kBS * kD;       // 1M elements
  char* p = (char*)d_ws;
  float* X32 = (float*)p;  p += NM * 4;     //  4 MB fp32 residual stream
  void*  X16 = (void*)p;   p += NM * 2;     //  2 MB f16 copy of X
  void*  E16 = (void*)p;   p += NM * 2;     //  2 MB f16 encoder_out
  void*  Qc  = (void*)p;   p += NM * 2;     //  2 MB f16 [row][h*DH+e]
  void*  Kc  = (void*)p;   p += NM * 2;
  void*  Vc  = (void*)p;   p += NM * 2;
  void*  Hc  = (void*)p;   p += NM * 2;     //  2 MB f16 head-concat output
  void*  H1  = (void*)p;   p += (size_t)kBS * kF * 2;            // 8 MB f16
  float* Ao  = (float*)p;  p += NM * 4;     //  4 MB fp32 sublayer output
  void*  P16 = (void*)p;   p += (size_t)kB * kH * kS * kS * 2;   // 16 MB f16 probs
  // fp32 attention scores live in d_out scratch; final GEMM overwrites it all.
  float* Sc = (float*)d_out;

  const int elemBlocks = (int)(NM / 256);

  embed_kernel<<<elemBlocks, 256, 0, stream>>>(seq, emb, pes, X32, X16, (int)NM);
  cvt16_kernel<<<elemBlocks, 256, 0, stream>>>(enc, E16, (int)NM);

  const long wqkvStride = (long)kH * kD * kDH;
  const long bqkvStride = (long)kH * kDH;
  const long sScore = (long)kS * kS;        // per-(b,h) score block

  for (int l = 0; l < kL; ++l) {
    // ---------------- Self-attention (causal) ----------------
    launch_gemm(stream, X16, wq_s + l * wqkvStride, Qc, bq_s + l * bqkvStride,
                kBS, kDH, kD, kD, kDH, kD, kH, kH,
                0, 0, 0, (long)kD * kDH, 0, kDH, kDH, 1.f, FLAG_OUTF16);
    launch_gemm(stream, X16, wk_s + l * wqkvStride, Kc, bk_s + l * bqkvStride,
                kBS, kDH, kD, kD, kDH, kD, kH, kH,
                0, 0, 0, (long)kD * kDH, 0, kDH, kDH, 1.f, FLAG_OUTF16);
    launch_gemm(stream, X16, wv_s + l * wqkvStride, Vc, bv_s + l * bqkvStride,
                kBS, kDH, kD, kD, kDH, kD, kH, kH,
                0, 0, 0, (long)kD * kDH, 0, kDH, kDH, 1.f, FLAG_OUTF16);
    launch_gemm(stream, Qc, Kc, Sc, nullptr,
                kS, kS, kDH, kD, kD, kS, kB * kH, kH,
                (long)kS * kD, kDH, (long)kS * kD, kDH,
                (long)kH * sScore, sScore, 0,
                0.125f, FLAG_TRANSB | FLAG_BF16);
    softmax_kernel<<<dim3(kS, kB * kH), 256, 0, stream>>>(Sc, P16, kS, kS, 1);
    launch_gemm(stream, P16, Vc, Hc, nullptr,
                kS, kDH, kS, kS, kD, kD, kB * kH, kH,
                (long)kH * sScore, sScore, (long)kS * kD, kDH,
                (long)kS * kD, kDH, 0, 1.f, FLAG_BF16 | FLAG_OUTF16);
    launch_gemm(stream, Hc, wo_s + (long)l * kD * kD, Ao, bo_s + (long)l * kD,
                kBS, kD, kD, kD, kD, kD, 1, 1, 0, 0, 0, 0, 0, 0, 0, 1.f, 0);
    bnres_kernel<<<elemBlocks, 256, 0, stream>>>(X32, X16, Ao, g1 + l * kD,
                                                 be1 + l * kD, m1 + l * kD,
                                                 v1 + l * kD, (int)NM);

    // ---------------- Cross-attention (encoder K/V) ----------------
    launch_gemm(stream, X16, wq_c + l * wqkvStride, Qc, bq_c + l * bqkvStride,
                kBS, kDH, kD, kD, kDH, kD, kH, kH,
                0, 0, 0, (long)kD * kDH, 0, kDH, kDH, 1.f, FLAG_OUTF16);
    launch_gemm(stream, E16, wk_c + l * wqkvStride, Kc, bk_c + l * bqkvStride,
                kB * kSE, kDH, kD, kD, kDH, kD, kH, kH,
                0, 0, 0, (long)kD * kDH, 0, kDH, kDH, 1.f, FLAG_OUTF16);
    launch_gemm(stream, E16, wv_c + l * wqkvStride, Vc, bv_c + l * bqkvStride,
                kB * kSE, kDH, kD, kD, kDH, kD, kH, kH,
                0, 0, 0, (long)kD * kDH, 0, kDH, kDH, 1.f, FLAG_OUTF16);
    launch_gemm(stream, Qc, Kc, Sc, nullptr,
                kS, kSE, kDH, kD, kD, kSE, kB * kH, kH,
                (long)kS * kD, kDH, (long)kSE * kD, kDH,
                (long)kH * sScore, sScore, 0,
                0.125f, FLAG_TRANSB | FLAG_BF16);
    softmax_kernel<<<dim3(kS, kB * kH), 256, 0, stream>>>(Sc, P16, kS, kSE, 0);
    launch_gemm(stream, P16, Vc, Hc, nullptr,
                kS, kDH, kSE, kSE, kD, kD, kB * kH, kH,
                (long)kH * sScore, sScore, (long)kSE * kD, kDH,
                (long)kS * kD, kDH, 0, 1.f, FLAG_BF16 | FLAG_OUTF16);
    launch_gemm(stream, Hc, wo_c + (long)l * kD * kD, Ao, bo_c + (long)l * kD,
                kBS, kD, kD, kD, kD, kD, 1, 1, 0, 0, 0, 0, 0, 0, 0, 1.f, 0);
    bnres_kernel<<<elemBlocks, 256, 0, stream>>>(X32, X16, Ao, g2 + l * kD,
                                                 be2 + l * kD, m2 + l * kD,
                                                 v2 + l * kD, (int)NM);

    // ---------------- FFN ----------------
    launch_gemm(stream, X16, w1 + (long)l * kD * kF, H1, b1 + (long)l * kF,
                kBS, kF, kD, kD, kF, kF, 1, 1, 0, 0, 0, 0, 0, 0, 0,
                1.f, FLAG_RELU | FLAG_OUTF16);
    launch_gemm(stream, H1, w2 + (long)l * kF * kD, Ao, b2 + (long)l * kD,
                kBS, kD, kF, kF, kD, kD, 1, 1, 0, 0, 0, 0, 0, 0, 0, 1.f, 0);
    bnres_kernel<<<elemBlocks, 256, 0, stream>>>(X32, X16, Ao, g3 + l * kD,
                                                 be3 + l * kD, m3 + l * kD,
                                                 v3 + l * kD, (int)NM);
  }

  // Final logits: [1024, D] x [D, V] + b_out -> d_out (fp32)
  launch_gemm(stream, X16, w_out, d_out, b_out,
              kBS, kV, kD, kD, kV, kV, 1, 1, 0, 0, 0, 0, 0, 0, 0, 1.f, 0);
}
